// EmbeddingGATEdge_13005160972437
// MI455X (gfx1250) — matrix operand order repaired and checked by reading, not compiled, not run on hardware
//
#include <hip/hip_runtime.h>
#include <hip/hip_bf16.h>

#define N_NODES   50000
#define N_EDGES   800000
#define EMBD      128
#define HID       32
#define HEADS     4
#define DDIM      128     // HID*HEADS
#define SLOPE     0.2f
#define LN_EPS    1e-5f

typedef __attribute__((ext_vector_type(16))) _Float16 v16h;
typedef __attribute__((ext_vector_type(2)))  _Float16 v2h;
typedef __attribute__((ext_vector_type(8)))  float    v8f;

// ---- monotonic float<->uint encoding for atomic max over floats ----
__device__ __forceinline__ unsigned enc_f32(float f) {
    unsigned u = __float_as_uint(f);
    return (u & 0x80000000u) ? ~u : (u | 0x80000000u);
}
__device__ __forceinline__ float dec_f32(unsigned u) {
    unsigned v = (u & 0x80000000u) ? (u & 0x7fffffffu) : ~u;
    return __uint_as_float(v);
}

// ---- transpose + convert fp32 weights (K x N, row-major) to f16 (N x K) ----
__global__ __launch_bounds__(256)
void k_convert_wT(const float* __restrict__ w, _Float16* __restrict__ wt) {
    int idx = blockIdx.x * 256 + threadIdx.x;   // 128*128 = 16384 elements
    int n = idx >> 7, k = idx & 127;
    wt[n * EMBD + k] = (_Float16)w[k * DDIM + n];
}

// ---- zero the accumulators / softmax state ----
__global__ __launch_bounds__(256)
void k_init(unsigned* __restrict__ mbuf, float* __restrict__ denom,
            float* __restrict__ acc) {
    int idx = blockIdx.x * 256 + threadIdx.x;
    if (idx < N_NODES * HEADS) { mbuf[idx] = 0u; denom[idx] = 0.f; }
    if (idx < N_NODES * DDIM)  acc[idx] = 0.f;
}

// ---- one 16x16 f32 output tile:   D += A(16x128 gathered f32->f16) * B(128x16 f16) ----
// A fragment layout (ISA 7.12.2, 16-bit A 16x32): lane group g=lane>>4, row=lane&15,
//   VGPR v holds K = (v<4?0:16) + g*8 + (v&3)*2  (low/high half = K, K+1)
// B fragment layout (16-bit B KxN=32x16): col = lane&15, VGPR v holds K = (v<4?0:16) + g*8 + (v&3)*2, +1
__device__ __forceinline__ v8f gemm_tile(const float* __restrict__ arow,
                                         const _Float16* __restrict__ wT,
                                         int n, int g) {
    v8f acc = {};
#pragma unroll
    for (int ks = 0; ks < EMBD; ks += 32) {
        v16h a, b;
#pragma unroll
        for (int v = 0; v < 8; ++v) {
            int kA = ks + ((v & 4) << 2) + g * 8 + (v & 3) * 2;
            float2 f2 = *reinterpret_cast<const float2*>(arow + kA);
            a[2 * v]     = (_Float16)f2.x;
            a[2 * v + 1] = (_Float16)f2.y;
            int kB = ks + ((v & 4) << 2) + g * 8 + (v & 3) * 2;
            v2h bp = *reinterpret_cast<const v2h*>(wT + n * EMBD + kB);
            b[2 * v]     = bp[0];
            b[2 * v + 1] = bp[1];
        }
        acc = __builtin_amdgcn_wmma_f32_16x16x32_f16(
            false, a, false, b, (short)0, acc, false, false);
    }
    return acc;
}

// ---- node pipeline: gather node_emb -> GEMM (WMMA) -> h, s_i, s_j ----
__global__ __launch_bounds__(256)
void k_node(const int* __restrict__ nf, const float* __restrict__ emb,
            const _Float16* __restrict__ wT, const float* __restrict__ bvec,
            const float* __restrict__ attn, float* __restrict__ h_out,
            float* __restrict__ s_i, float* __restrict__ s_j) {
    __shared__ float tile[16 * 132];
    int tid = threadIdx.x;
    int lane = tid & 31, wave = tid >> 5;
    int g = lane >> 4, mr = lane & 15;
    int m0 = blockIdx.x * 16;                 // 50000 % 16 == 0
    int row = nf[m0 + mr];
    const float* arow = emb + (long)row * EMBD;
    int n = wave * 16 + mr;
    v8f acc = gemm_tile(arow, wT, n, g);
    float bias = bvec[n];
#pragma unroll
    for (int r = 0; r < 8; ++r)
        tile[(g * 8 + r) * 132 + n] = acc[r] + bias;
    __syncthreads();
    // write h rows (coalesced)
#pragma unroll
    for (int k2 = 0; k2 < 8; ++k2) {
        int idx = tid + 256 * k2;
        int r = idx >> 7, c = idx & 127;
        h_out[(long)(m0 + r) * DDIM + c] = tile[r * 132 + c];
    }
    // s_i / s_j: 16 rows x 4 heads x {i,j}
    if (tid < 128) {
        int r = tid >> 3, rem = tid & 7, head = rem >> 1, which = rem & 1;
        const float* av = attn + head * (3 * HID) + which * HID;
        const float* trow = tile + r * 132 + head * HID;
        float s = 0.f;
#pragma unroll
        for (int f = 0; f < HID; ++f) s += trow[f] * av[f];
        float* dst = which ? s_j : s_i;
        dst[(m0 + r) * HEADS + head] = s;
    }
}

// ---- edge pipeline: gather edge_emb -> GEMM (WMMA) -> s_e + LayerNorm(eh2) in-LDS ----
__global__ __launch_bounds__(256)
void k_edge(const int* __restrict__ ef, const float* __restrict__ emb,
            const _Float16* __restrict__ wT, const float* __restrict__ bvec,
            const float* __restrict__ attn, const float* __restrict__ gamma,
            const float* __restrict__ beta, float* __restrict__ s_e,
            float* __restrict__ eh2) {
    __shared__ float tile[16 * 132];
    __shared__ float red_s[256];
    __shared__ float red_q[256];
    __shared__ float mu_s[16], ri_s[16];
    int tid = threadIdx.x;
    int lane = tid & 31, wave = tid >> 5;
    int g = lane >> 4, mr = lane & 15;
    long e0 = (long)blockIdx.x * 16;          // 800000 % 16 == 0
    int row = ef[e0 + mr];
    const float* arow = emb + (long)row * EMBD;
    int n = wave * 16 + mr;
    v8f acc = gemm_tile(arow, wT, n, g);
    float bias = bvec[n];
#pragma unroll
    for (int r = 0; r < 8; ++r)
        tile[(g * 8 + r) * 132 + n] = acc[r] + bias;
    __syncthreads();
    // s_e = <eh[row, head, :], a_e[head, :]>
    if (tid < 64) {
        int r = tid >> 2, head = tid & 3;
        const float* av = attn + head * (3 * HID) + 2 * HID;
        const float* trow = tile + r * 132 + head * HID;
        float s = 0.f;
#pragma unroll
        for (int f = 0; f < HID; ++f) s += trow[f] * av[f];
        s_e[(e0 + r) * HEADS + head] = s;
    }
    // LayerNorm partials: 16 threads per row, 8 cols each
    {
        int r = tid >> 4, l = tid & 15;
        float s = 0.f, q = 0.f;
#pragma unroll
        for (int j = 0; j < 8; ++j) {
            float v = tile[r * 132 + l + 16 * j];
            s += v; q += v * v;
        }
        red_s[tid] = s; red_q[tid] = q;
    }
    __syncthreads();
    if (tid < 16) {
        float s = 0.f, q = 0.f;
        for (int i = 0; i < 16; ++i) { s += red_s[tid * 16 + i]; q += red_q[tid * 16 + i]; }
        float mu = s * (1.f / 128.f);
        float var = q * (1.f / 128.f) - mu * mu;
        mu_s[tid] = mu;
        ri_s[tid] = rsqrtf(var + LN_EPS);
    }
    __syncthreads();
#pragma unroll
    for (int k2 = 0; k2 < 8; ++k2) {
        int idx = tid + 256 * k2;
        int r = idx >> 7, c = idx & 127;
        float v = (tile[r * 132 + c] - mu_s[r]) * ri_s[r] * gamma[c] + beta[c];
        __builtin_nontemporal_store(v, eh2 + (e0 + r) * DDIM + c);  // 410 MB stream
    }
}

// ---- per (edge, head): leaky_relu score + segment max via atomic max ----
__global__ __launch_bounds__(256)
void k_score(const int* __restrict__ tgt, const int* __restrict__ src,
             const float* __restrict__ s_i, const float* __restrict__ s_j,
             const float* __restrict__ s_e, float* __restrict__ eij,
             unsigned* __restrict__ mbuf) {
    int idx = blockIdx.x * 256 + threadIdx.x;
    if (idx >= N_EDGES * HEADS) return;
    int e = idx >> 2, head = idx & 3;
    int t = tgt[e], s = src[e];
    float x = s_i[t * HEADS + head] + s_j[s * HEADS + head] + s_e[idx];
    x = x > 0.f ? x : SLOPE * x;
    eij[idx] = x;
    atomicMax(mbuf + t * HEADS + head, enc_f32(x));
}

// ---- ex = exp(eij - m[tgt]); denom += ex ----
__global__ __launch_bounds__(256)
void k_expsum(const int* __restrict__ tgt, const unsigned* __restrict__ mbuf,
              float* __restrict__ eij, float* __restrict__ denom) {
    int idx = blockIdx.x * 256 + threadIdx.x;
    if (idx >= N_EDGES * HEADS) return;
    int e = idx >> 2, head = idx & 3;
    int t = tgt[e];
    float ex = expf(eij[idx] - dec_f32(mbuf[t * HEADS + head]));
    eij[idx] = ex;
    __hip_atomic_fetch_add(&denom[t * HEADS + head], ex,
                           __ATOMIC_RELAXED, __HIP_MEMORY_SCOPE_AGENT);
}

// ---- acc[tgt] += alpha * h[src]; one wave32 per edge, 4 cols per lane ----
__global__ __launch_bounds__(256)
void k_message(const int* __restrict__ tgt, const int* __restrict__ src,
               const float* __restrict__ ex, const float* __restrict__ denom,
               const float* __restrict__ h, float* __restrict__ acc) {
    int lane = threadIdx.x & 31;
    long e = (long)blockIdx.x * 8 + (threadIdx.x >> 5);
    if (e >= N_EDGES) return;
    int t = tgt[e], s = src[e];
#pragma unroll
    for (int k = 0; k < 4; ++k) {
        float alpha = ex[e * HEADS + k] / denom[t * HEADS + k];
        int c = k * 32 + lane;
        float hv = h[(long)s * DDIM + c];
        __hip_atomic_fetch_add(&acc[(long)t * DDIM + c], alpha * hv,
                               __ATOMIC_RELAXED, __HIP_MEMORY_SCOPE_AGENT);
    }
}

// ---- out = LayerNorm(acc + h); one wave32 per node, shuffle reduction ----
__global__ __launch_bounds__(256)
void k_final(const float* __restrict__ acc, const float* __restrict__ h,
             const float* __restrict__ gamma, const float* __restrict__ beta,
             float* __restrict__ out) {
    int lane = threadIdx.x & 31;
    long node = (long)blockIdx.x * 8 + (threadIdx.x >> 5);
    if (node >= N_NODES) return;
    float v[4];
    float s = 0.f;
#pragma unroll
    for (int k = 0; k < 4; ++k) {
        long i = node * DDIM + k * 32 + lane;
        v[k] = acc[i] + h[i];
        s += v[k];
    }
#pragma unroll
    for (int off = 16; off >= 1; off >>= 1) s += __shfl_xor(s, off);
    float mu = s * (1.f / 128.f);
    float q = 0.f;
#pragma unroll
    for (int k = 0; k < 4; ++k) { float d = v[k] - mu; q += d * d; }
#pragma unroll
    for (int off = 16; off >= 1; off >>= 1) q += __shfl_xor(q, off);
    float rinv = rsqrtf(q * (1.f / 128.f) + LN_EPS);
#pragma unroll
    for (int k = 0; k < 4; ++k) {
        int c = k * 32 + lane;
        float o = (v[k] - mu) * rinv * gamma[c] + beta[c];
        __builtin_nontemporal_store(o, out + node * DDIM + c);
    }
}

extern "C" void kernel_launch(void* const* d_in, const int* in_sizes, int n_in,
                              void* d_out, int out_size, void* d_ws, size_t ws_size,
                              hipStream_t stream) {
    (void)in_sizes; (void)n_in; (void)out_size; (void)ws_size;
    const int*   node_features = (const int*)d_in[0];
    const int*   edge_features = (const int*)d_in[1];
    const int*   edge_index    = (const int*)d_in[2];
    const float* node_emb      = (const float*)d_in[3];
    const float* edge_emb      = (const float*)d_in[4];
    const float* w_W           = (const float*)d_in[5];
    const float* w_b           = (const float*)d_in[6];
    const float* ew_W          = (const float*)d_in[7];
    const float* ew_b          = (const float*)d_in[8];
    const float* attn          = (const float*)d_in[9];
    const float* ln_gamma      = (const float*)d_in[10];
    const float* ln_beta       = (const float*)d_in[11];
    const int* tgt = edge_index;                 // edge_index[0]
    const int* src = edge_index + N_EDGES;       // edge_index[1]

    // workspace carve-up (~80 MB)
    char* ws = (char*)d_ws;
    size_t off = 0;
    auto carve = [&](size_t bytes) -> char* {
        char* p = ws + off;
        off += (bytes + 255) & ~(size_t)255;
        return p;
    };
    _Float16* wT    = (_Float16*)carve((size_t)EMBD * DDIM * 2);
    _Float16* ewT   = (_Float16*)carve((size_t)EMBD * DDIM * 2);
    float*    h     = (float*)carve((size_t)N_NODES * DDIM * 4);
    float*    s_i   = (float*)carve((size_t)N_NODES * HEADS * 4);
    float*    s_j   = (float*)carve((size_t)N_NODES * HEADS * 4);
    float*    s_e   = (float*)carve((size_t)N_EDGES * HEADS * 4);
    float*    eij   = (float*)carve((size_t)N_EDGES * HEADS * 4);
    unsigned* mbuf  = (unsigned*)carve((size_t)N_NODES * HEADS * 4);
    float*    denom = (float*)carve((size_t)N_NODES * HEADS * 4);
    float*    accb  = (float*)carve((size_t)N_NODES * DDIM * 4);

    float* out_nodes = (float*)d_out;
    float* out_eh2   = out_nodes + (size_t)N_NODES * DDIM;

    k_convert_wT<<<64, 256, 0, stream>>>(w_W, wT);
    k_convert_wT<<<64, 256, 0, stream>>>(ew_W, ewT);
    k_init<<<(N_NODES * DDIM + 255) / 256, 256, 0, stream>>>(mbuf, denom, accb);
    k_node<<<N_NODES / 16, 256, 0, stream>>>(node_features, node_emb, wT, w_b,
                                             attn, h, s_i, s_j);
    k_edge<<<N_EDGES / 16, 256, 0, stream>>>(edge_features, edge_emb, ewT, ew_b,
                                             attn, ln_gamma, ln_beta, s_e, out_eh2);
    k_score<<<(N_EDGES * HEADS + 255) / 256, 256, 0, stream>>>(tgt, src, s_i, s_j,
                                                               s_e, eij, mbuf);
    k_expsum<<<(N_EDGES * HEADS + 255) / 256, 256, 0, stream>>>(tgt, mbuf, eij, denom);
    k_message<<<N_EDGES / 8, 256, 0, stream>>>(tgt, src, eij, denom, h, accb);
    k_final<<<(N_NODES + 7) / 8, 256, 0, stream>>>(accb, h, ln_gamma, ln_beta, out_nodes);
}
